// SwitchableQATGPT2Attention_65429531787888
// MI455X (gfx1250) — matrix operand order repaired and checked by reading, not compiled, not run on hardware
//
#include <hip/hip_runtime.h>
#include <hip/hip_bf16.h>

// ---------------- types ----------------
typedef __attribute__((ext_vector_type(16))) __bf16 v16bf;
typedef __attribute__((ext_vector_type(4)))  __bf16 v4bf;
typedef __attribute__((ext_vector_type(8)))  float  v8f;

union FragU {
    v16bf v;
    uint2 d2[4];
    uint4 d4[2];
};

__device__ __forceinline__ __bf16 f2bf(float f) { return (__bf16)f; }

__device__ __forceinline__ v8f wmma_bf16(v16bf a, v16bf b, v8f c) {
    return __builtin_amdgcn_wmma_f32_16x16x32_bf16(false, a, false, b, (short)0, c, false, false);
}

// A-matrix fragment (16x32 bf16, MxK), row-major source with leading dim `ld` elems.
// Lane L: row = L&15, kb8 = (L>>4)*8. Elems 0..15 = K 2*kb8..2*kb8+15 (contiguous).
__device__ __forceinline__ v16bf fragA_rowmajor(const __bf16* base, int row, int ld, int kb8) {
    FragU f;
    const __bf16* p = base + row * ld + 2 * kb8;
    f.d2[0] = *(const uint2*)(p);
    f.d2[1] = *(const uint2*)(p + 4);
    f.d2[2] = *(const uint2*)(p + 8);
    f.d2[3] = *(const uint2*)(p + 12);
    return f.v;
}

// B-matrix fragment (32x16 bf16, KxN): lane L holds col L&15, K rows (L>>4)*16..+15
// contiguous. `p` points at the 16 contiguous K elements for this lane.
__device__ __forceinline__ v16bf fragB_contig_lds(const __bf16* p) {
    FragU f;
    f.d2[0] = *(const uint2*)(p);
    f.d2[1] = *(const uint2*)(p + 4);
    f.d2[2] = *(const uint2*)(p + 8);
    f.d2[3] = *(const uint2*)(p + 12);
    return f.v;
}

__device__ __forceinline__ v16bf frag_contig_g16(const __bf16* p) {
    FragU f;
    f.d4[0] = *(const uint4*)(p);
    f.d4[1] = *(const uint4*)(p + 8);
    return f.v;
}

// ---------------- constants ----------------
#define TT   2048
#define CC   1024
#define HH   16
#define DD   64
#define MM   4096      // B*T
#define KDIM 1024
#define LR   16        // lora rank
#define BMT  64
#define BNT  64
#define BKT  32
#define LDT  40        // LDS leading dim (elems): 80B rows -> 8B-aligned fragments

// ---------------- kernel 1: lora intermediate t = 2*(X @ LA), bf16 ----------------
__global__ __launch_bounds__(256)
void lora_t_kernel(const float* __restrict__ X, const float* __restrict__ LA,
                   __bf16* __restrict__ Tl) {
    int id = blockIdx.x * 256 + threadIdx.x;     // MM*16 ids
    if (id >= MM * LR) return;
    int row = id >> 4, rr = id & 15;
    const float* xr = X + (size_t)row * KDIM;
    float acc = 0.f;
#pragma unroll 8
    for (int k = 0; k < KDIM; ++k) acc += xr[k] * LA[k * LR + rr];
    Tl[id] = f2bf(2.0f * acc);                   // LORA_SCALE = 32/16
}

// ---------------- kernel 2/5: GEMM  out = A@W + bias + Tl@LB ----------------
// mode 0: qkv epilogue (split + fake-quant k/v, scale q by 0.125, bf16 stores)
// mode 1: plain fp32 store to outF
__global__ __launch_bounds__(128)
void lora_gemm_kernel(const float* __restrict__ A, const float* __restrict__ W,
                      const __bf16* __restrict__ Tl, const float* __restrict__ LB,
                      const float* __restrict__ bias, int N, int mode,
                      float* __restrict__ outF,
                      __bf16* __restrict__ Qb, __bf16* __restrict__ Kb,
                      __bf16* __restrict__ Vt,
                      const float* __restrict__ kv_scale_p,
                      const float* __restrict__ kv_zp_p) {
    __shared__ __bf16 sA[2 * BMT * LDT];
    __shared__ __bf16 sB[2 * BNT * LDT];   // W tile transposed: sB[n][k]

    const int tid  = threadIdx.x;
    const int m0   = blockIdx.y * BMT;
    const int n0   = blockIdx.x * BNT;
    const int wave = tid >> 5, lane = tid & 31;
    const int wm   = (wave >> 1) * 32, wn = (wave & 1) * 32;
    const int ln15 = lane & 15, half = lane >> 4, kb8 = half * 8;

    // staging decomposition (branch-free, vectorized)
    const int arow = tid >> 3, akc = (tid & 7) * 4;    // +p*16 rows, float4 along K
    const int bkk  = tid >> 4, bnc = (tid & 15) * 4;   // +p*8  k-rows, float4 along N

    v8f acc[2][2];
#pragma unroll
    for (int i = 0; i < 2; ++i)
#pragma unroll
        for (int j = 0; j < 2; ++j) acc[i][j] = (v8f){};

    float4 ra[4], rb[4];

    auto gload = [&](int k0) {
#pragma unroll
        for (int p = 0; p < 4; ++p)
            ra[p] = *(const float4*)&A[(size_t)(m0 + arow + p * 16) * KDIM + k0 + akc];
#pragma unroll
        for (int p = 0; p < 4; ++p)
            rb[p] = *(const float4*)&W[(size_t)(k0 + bkk + p * 8) * N + n0 + bnc];
    };
    auto sstore = [&](int buf) {
        __bf16* dA = sA + buf * (BMT * LDT);
        __bf16* dB = sB + buf * (BNT * LDT);
#pragma unroll
        for (int p = 0; p < 4; ++p) {
            v4bf t;
            t[0] = f2bf(ra[p].x); t[1] = f2bf(ra[p].y);
            t[2] = f2bf(ra[p].z); t[3] = f2bf(ra[p].w);
            *(v4bf*)(dA + (arow + p * 16) * LDT + akc) = t;
        }
#pragma unroll
        for (int p = 0; p < 4; ++p) {       // transposed scatter: sB[n][k]
            int kk = bkk + p * 8;
            dB[(bnc + 0) * LDT + kk] = f2bf(rb[p].x);
            dB[(bnc + 1) * LDT + kk] = f2bf(rb[p].y);
            dB[(bnc + 2) * LDT + kk] = f2bf(rb[p].z);
            dB[(bnc + 3) * LDT + kk] = f2bf(rb[p].w);
        }
    };
    auto compute = [&](int buf) {
        const __bf16* bA = sA + buf * (BMT * LDT);
        const __bf16* bB = sB + buf * (BNT * LDT);
        v16bf a0 = fragA_rowmajor(bA, wm + ln15,      LDT, kb8);
        v16bf a1 = fragA_rowmajor(bA, wm + 16 + ln15, LDT, kb8);
        v16bf b0 = fragB_contig_lds(bB + (wn + ln15)      * LDT + half * 16);
        v16bf b1 = fragB_contig_lds(bB + (wn + 16 + ln15) * LDT + half * 16);
        acc[0][0] = wmma_bf16(a0, b0, acc[0][0]);
        acc[0][1] = wmma_bf16(a0, b1, acc[0][1]);
        acc[1][0] = wmma_bf16(a1, b0, acc[1][0]);
        acc[1][1] = wmma_bf16(a1, b1, acc[1][1]);
    };

    const int NS = KDIM / BKT;   // 32
    gload(0);
    sstore(0);
    __syncthreads();
    for (int s = 0; s < NS; ++s) {
        if (s + 1 < NS) gload((s + 1) * BKT);    // overlap with WMMA below
        compute(s & 1);
        if (s + 1 < NS) sstore((s + 1) & 1);
        __syncthreads();
    }

    // ---- LoRA K-step (rank 16, zero-padded to K=32), staged into buffer 0 ----
    {
        __bf16* dA = sA;
        __bf16* dB = sB;
#pragma unroll
        for (int p = 0; p < 4; ++p) {
            int row = arow + p * 16;
            uint2 z = {0u, 0u};
            uint2 v = (akc < LR) ? *(const uint2*)&Tl[(size_t)(m0 + row) * LR + akc] : z;
            *(uint2*)(dA + row * LDT + akc) = v;    // Tl already bf16 + prescaled
        }
#pragma unroll
        for (int p = 0; p < 2; ++p) {               // kk in [0,16): real LB rows
            int kk = bkk + p * 8;
            float4 v = *(const float4*)&LB[(size_t)kk * N + n0 + bnc];
            dB[(bnc + 0) * LDT + kk] = f2bf(v.x);
            dB[(bnc + 1) * LDT + kk] = f2bf(v.y);
            dB[(bnc + 2) * LDT + kk] = f2bf(v.z);
            dB[(bnc + 3) * LDT + kk] = f2bf(v.w);
        }
#pragma unroll
        for (int p = 2; p < 4; ++p) {               // kk in [16,32): zero pad
            int kk = bkk + p * 8;
            __bf16 z = f2bf(0.f);
            dB[(bnc + 0) * LDT + kk] = z;
            dB[(bnc + 1) * LDT + kk] = z;
            dB[(bnc + 2) * LDT + kk] = z;
            dB[(bnc + 3) * LDT + kk] = z;
        }
        __syncthreads();
        compute(0);
    }

    // ---- epilogue ----
    const float sc = (mode == 0) ? *kv_scale_p : 1.f;
    const float zp = (mode == 0) ? *kv_zp_p : 0.f;

#pragma unroll
    for (int mt = 0; mt < 2; ++mt)
#pragma unroll
        for (int nt = 0; nt < 2; ++nt)
#pragma unroll
            for (int i = 0; i < 8; ++i) {
                int mg = m0 + wm + mt * 16 + i + 8 * half;
                int ng = n0 + wn + nt * 16 + ln15;
                float val = acc[mt][nt][i] + bias[ng];
                if (mode == 1) {
                    outF[(size_t)mg * N + ng] = val;
                } else {
                    int regn = ng >> 10;
                    int h = (ng & 1023) >> 6;
                    int d = ng & 63;
                    int b = mg >> 11;
                    int t = mg & (TT - 1);
                    size_t bh = (size_t)b * HH + h;
                    if (regn == 0) {
                        Qb[(bh * TT + t) * DD + d] = f2bf(val * 0.125f);  // fold 1/sqrt(64)
                    } else {
                        float q = rintf(val / sc + zp);                    // round-half-even
                        q = fminf(fmaxf(q, 0.f), 255.f);
                        float fq = (q - zp) * sc;
                        if (regn == 1) Kb[(bh * TT + t) * DD + d] = f2bf(fq);
                        else           Vt[(bh * DD + d) * TT + t] = f2bf(fq);  // V transposed
                    }
                }
            }
}

// ---------------- kernel 3: flash attention, 1 wave per (b,h, 16-query tile) -------
__global__ __launch_bounds__(32)
void flash_attn_kernel(const __bf16* __restrict__ Qb, const __bf16* __restrict__ Kb,
                       const __bf16* __restrict__ Vt, float* __restrict__ Oout) {
    __shared__ __bf16 sP[16 * LDT];

    const int blk = blockIdx.x;
    const int bh = blk >> 7;        // B*H = 32
    const int qt = blk & 127;       // T/16 = 128
    const int b = bh >> 4, h = bh & 15;
    const __bf16* Qh = Qb + (size_t)bh * TT * DD;
    const __bf16* Kh = Kb + (size_t)bh * TT * DD;
    const __bf16* Vh = Vt + (size_t)bh * DD * TT;

    const int lane = threadIdx.x;
    const int ln15 = lane & 15, half = lane >> 4, kb8 = half * 8;
    const int q0 = qt * 16;

    // Q fragments (A-layout): two K=32 chunks along D; per-lane row is contiguous,
    // 16 elems at 2*kb8 as two 16B loads.
    v16bf qf[2];
#pragma unroll
    for (int ch = 0; ch < 2; ++ch) {
        FragU f;
        const __bf16* p = Qh + (size_t)(q0 + ln15) * DD + ch * 32;
        f.d4[0] = *(const uint4*)(p + 2 * kb8);
        f.d4[1] = *(const uint4*)(p + 2 * kb8 + 8);
        qf[ch] = f.v;
    }

    float m_i[8], l_i[8];
    v8f o[4];
#pragma unroll
    for (int i = 0; i < 8; ++i) { m_i[i] = -1e30f; l_i[i] = 0.f; }
#pragma unroll
    for (int nt = 0; nt < 4; ++nt) o[nt] = (v8f){};

    const int nkb = ((q0 + 15) >> 5) + 1;          // causal: 32-key blocks
    for (int kb = 0; kb < nkb; ++kb) {
        const int kbase = kb * 32;
        v8f s[2] = {(v8f){}, (v8f){}};
#pragma unroll
        for (int t = 0; t < 2; ++t) {
#pragma unroll
            for (int ch = 0; ch < 2; ++ch) {
                v16bf kf = frag_contig_g16(
                    Kh + (size_t)(kbase + t * 16 + ln15) * DD + ch * 32 + half * 16);
                s[t] = wmma_bf16(qf[ch], kf, s[t]);
            }
        }
        // causal mask
#pragma unroll
        for (int t = 0; t < 2; ++t) {
            int key = kbase + t * 16 + ln15;
#pragma unroll
            for (int i = 0; i < 8; ++i) {
                int qrow = q0 + i + 8 * half;
                if (key > qrow) s[t][i] = -1e30f;
            }
        }
        // row max across the 16 lanes of each half (wave32: xor<=8 stays in-half)
        float mnew[8], alpha[8];
#pragma unroll
        for (int i = 0; i < 8; ++i) {
            float mx = fmaxf(s[0][i], s[1][i]);
            mx = fmaxf(mx, __shfl_xor(mx, 1));
            mx = fmaxf(mx, __shfl_xor(mx, 2));
            mx = fmaxf(mx, __shfl_xor(mx, 4));
            mx = fmaxf(mx, __shfl_xor(mx, 8));
            mnew[i] = fmaxf(m_i[i], mx);
            alpha[i] = __expf(m_i[i] - mnew[i]);
        }
        // P = exp(S - mnew): C-layout -> LDS (row-major 16x32) + row sums
        float rsum[8];
#pragma unroll
        for (int i = 0; i < 8; ++i) rsum[i] = 0.f;
#pragma unroll
        for (int t = 0; t < 2; ++t)
#pragma unroll
            for (int i = 0; i < 8; ++i) {
                float pv = __expf(s[t][i] - mnew[i]);
                sP[(i + 8 * half) * LDT + t * 16 + ln15] = f2bf(pv);
                rsum[i] += pv;
            }
#pragma unroll
        for (int i = 0; i < 8; ++i) {
            float rs = rsum[i];
            rs += __shfl_xor(rs, 1);
            rs += __shfl_xor(rs, 2);
            rs += __shfl_xor(rs, 4);
            rs += __shfl_xor(rs, 8);
            l_i[i] = l_i[i] * alpha[i] + rs;
            m_i[i] = mnew[i];
        }
#pragma unroll
        for (int nt = 0; nt < 4; ++nt)
#pragma unroll
            for (int i = 0; i < 8; ++i) o[nt][i] *= alpha[i];

        __syncthreads();
        v16bf pf = fragA_rowmajor(sP, ln15, LDT, kb8);   // P as A-matrix 16x32
#pragma unroll
        for (int nt = 0; nt < 4; ++nt) {
            v16bf vf = frag_contig_g16(
                Vh + (size_t)(nt * 16 + ln15) * TT + kbase + half * 16);
            o[nt] = wmma_bf16(pf, vf, o[nt]);
        }
        __syncthreads();
    }

    // normalize + write [B,T,C] fp32
#pragma unroll
    for (int i = 0; i < 8; ++i) l_i[i] = 1.0f / l_i[i];
#pragma unroll
    for (int nt = 0; nt < 4; ++nt)
#pragma unroll
        for (int i = 0; i < 8; ++i) {
            int qrow = q0 + i + 8 * half;
            int d = nt * 16 + ln15;
            Oout[((size_t)b * TT + qrow) * CC + h * DD + d] = o[nt][i] * l_i[i];
        }
}

// ---------------- launch ----------------
extern "C" void kernel_launch(void* const* d_in, const int* in_sizes, int n_in,
                              void* d_out, int out_size, void* d_ws, size_t ws_size,
                              hipStream_t stream) {
    (void)in_sizes; (void)n_in; (void)out_size; (void)ws_size;
    const float* x        = (const float*)d_in[0];
    const float* w_attn   = (const float*)d_in[1];
    const float* b_attn   = (const float*)d_in[2];
    const float* la_attn  = (const float*)d_in[3];
    const float* lb_attn  = (const float*)d_in[4];
    const float* w_proj   = (const float*)d_in[5];
    const float* b_proj   = (const float*)d_in[6];
    const float* la_proj  = (const float*)d_in[7];
    const float* lb_proj  = (const float*)d_in[8];
    const float* kv_scale = (const float*)d_in[9];
    const float* kv_zp    = (const float*)d_in[10];

    char* ws = (char*)d_ws;
    __bf16* tA = (__bf16*)(ws);                                   // 128 KB
    __bf16* Qb = (__bf16*)(ws + 0x20000);                         // 8 MB
    __bf16* Kb = (__bf16*)(ws + 0x20000 + 0x800000);              // 8 MB
    __bf16* Vt = (__bf16*)(ws + 0x20000 + 2 * 0x800000);          // 8 MB
    float*  AO = (float*) (ws + 0x20000 + 3 * 0x800000);          // 16 MB
    __bf16* tP = (__bf16*)(ws + 0x20000 + 3 * 0x800000 + 0x1000000);

    // 1) lora t for qkv
    lora_t_kernel<<<(MM * LR + 255) / 256, 256, 0, stream>>>(x, la_attn, tA);
    // 2) qkv GEMM + split + fake quant
    lora_gemm_kernel<<<dim3(3 * CC / BNT, MM / BMT), 128, 0, stream>>>(
        x, w_attn, tA, lb_attn, b_attn, 3 * CC, 0,
        nullptr, Qb, Kb, Vt, kv_scale, kv_zp);
    // 3) flash attention
    flash_attn_kernel<<<dim3(2 * HH * (TT / 16)), 32, 0, stream>>>(Qb, Kb, Vt, AO);
    // 4) lora t for proj
    lora_t_kernel<<<(MM * LR + 255) / 256, 256, 0, stream>>>(AO, la_proj, tP);
    // 5) output projection
    lora_gemm_kernel<<<dim3(CC / BNT, MM / BMT), 128, 0, stream>>>(
        AO, w_proj, tP, lb_proj, b_proj, CC, 1,
        (float*)d_out, nullptr, nullptr, nullptr, kv_scale, kv_zp);
}
